// GroupInvariance_1906965479549
// MI455X (gfx1250) — matrix operand order repaired and checked by reading, not compiled, not run on hardware
//
#include <hip/hip_runtime.h>

typedef __attribute__((ext_vector_type(16))) _Float16 v16h;
typedef __attribute__((ext_vector_type(8)))  _Float16 v8h;
typedef __attribute__((ext_vector_type(8)))  float    v8f;

#define BS               262144
#define NPER             5
#define ROWS_PER_GROUP   16
#define NGROUPS          (BS / ROWS_PER_GROUP)   // 16384 wave-tasks
#define WAVES_PER_BLOCK  8
#define BLOCK_THREADS    256
#define NUM_BLOCKS       1024                     // 8192 waves -> 2 groups each

#define LOG2E  1.4426950408889634f
#define TWO_LOG2E (2.0f * LOG2E)

// v_exp_f32 computes 2^x; v_rcp_f32 is a single TRANS op.
__device__ __forceinline__ float exp2_fast(float x) { return __builtin_amdgcn_exp2f(x); }
__device__ __forceinline__ float rcp_fast(float x)  { return __builtin_amdgcn_rcpf(x); }

// tanh(y) given pre-scaled arg z = 2*log2e*y:  tanh = 1 - 2/(2^z + 1)
__device__ __forceinline__ float tanh_scaled(float z) {
    return fmaf(-2.0f, rcp_fast(exp2_fast(z) + 1.0f), 1.0f);
}
// sigmoid(y) given pre-scaled arg z = -log2e*y:  sig = 1/(2^z + 1)
__device__ __forceinline__ float sigmoid_scaled(float z) {
    return rcp_fast(exp2_fast(z) + 1.0f);
}

__global__ void __launch_bounds__(BLOCK_THREADS)
group_invariance_fused(const float* __restrict__ inputs,
                       const float* __restrict__ W1,  const float* __restrict__ b1,
                       const float* __restrict__ W2,  const float* __restrict__ b2,
                       const float* __restrict__ W3,  const float* __restrict__ b3,
                       const float* __restrict__ Wfc, const float* __restrict__ Wout,
                       const float* __restrict__ bout,
                       float* __restrict__ out)
{
    // per-wave LDS regions (wave-local: DS ops are in-order per wave)
    __shared__ _Float16 h2buf_all[WAVES_PER_BLOCK][16 * 64];       // 16 KB
    __shared__ float    sbuf_all [WAVES_PER_BLOCK][NPER][16 * 10]; // 25.6 KB
    __shared__ float    ybuf_all [WAVES_PER_BLOCK][16 * 2];        // 1 KB

    const int lane        = threadIdx.x & 31;
    const int waveInBlock = threadIdx.x >> 5;
    const int h           = lane >> 4;    // lane half (0/1)
    const int m           = lane & 15;    // row-in-group (A row) / column index (B/D col)

    _Float16* h2buf        = h2buf_all[waveInBlock];
    float (*sbuf)[16 * 10] = sbuf_all[waveInBlock];
    float* ybuf            = ybuf_all[waveInBlock];

    // ---------------- hoisted per-lane weight fragments ----------------
    // layer1: this lane's A-slots i=0..7 need K = 8*h + i.
    // Pre-scale by 2*log2e so the activation prep is a single FMA feeding exp2.
    float w1s[8], b1s[8];
    #pragma unroll
    for (int j = 0; j < 8; ++j) {
        w1s[j] = W1[8 * h + j] * TWO_LOG2E;
        b1s[j] = b1[8 * h + j] * TWO_LOG2E;
    }

    // layer2 B fragments (W2: 16x64 row-major). B slot i, half h -> K = 8*h + i (i<8), 16 + 8*h + (i-8) (i>=8) (pad K 16->32)
    v16h  b2frag[4];
    float b2s[4];                                  // bias pre-scaled by 2*log2e
    #pragma unroll
    for (int nt = 0; nt < 4; ++nt) {
        const int col = nt * 16 + m;
        b2s[nt] = b2[col] * TWO_LOG2E;
        #pragma unroll
        for (int i = 0; i < 16; ++i) {
            const int K = (i < 8) ? (8 * h + i) : (16 + 8 * h + (i - 8));
            b2frag[nt][i] = (K < 16) ? (_Float16)W2[K * 64 + col] : (_Float16)0.0f;
        }
    }

    // layer3 B fragments (W3: 64x10 row-major), cols 10..15 zero-padded; 2 K-tiles of 32
    v16h b3frag[2];
    const float b3s = (m < 10) ? (-LOG2E * b3[m]) : 0.0f;   // bias pre-scaled by -log2e
    #pragma unroll
    for (int kt = 0; kt < 2; ++kt) {
        #pragma unroll
        for (int i = 0; i < 16; ++i) {
            const int K = kt * 32 + ((i < 8) ? (8 * h + i) : (16 + 8 * h + (i - 8)));
            b3frag[kt][i] = (m < 10) ? (_Float16)W3[K * 10 + m] : (_Float16)0.0f;
        }
    }

    const int waveGlobal = blockIdx.x * WAVES_PER_BLOCK + waveInBlock;
    const int nWaves     = NUM_BLOCKS * WAVES_PER_BLOCK;

    for (int g = waveGlobal; g < NGROUPS; g += nWaves) {
        const int rowBase = g * ROWS_PER_GROUP;

        #pragma unroll
        for (int n = 0; n < NPER; ++n) {
            // ---- layer1: scalar -> 16 features, packed straight into an A-fragment
            const float t = inputs[(rowBase + m) * NPER + n];
            v16h a1;
            #pragma unroll
            for (int i = 0; i < 16; ++i) {
                if (i < 8) {
                    a1[i] = (_Float16)tanh_scaled(fmaf(t, w1s[i], b1s[i]));
                } else {
                    a1[i] = (_Float16)0.0f;   // K 16..31 pad
                }
            }

            // ---- layer2: (16x16, K=16 padded to 32) x (16x64) -> 4 WMMAs
            v8f c2[4];
            #pragma unroll
            for (int nt = 0; nt < 4; ++nt) {
                v8f z = {};
                c2[nt] = __builtin_amdgcn_wmma_f32_16x16x32_f16(
                    false, a1, false, b2frag[nt], (short)0, z, false, false);
            }

            // bias + tanh (one FMA of prep), deposit h2 tile (16x64 f16, row-major) to LDS
            #pragma unroll
            for (int nt = 0; nt < 4; ++nt) {
                const int col = nt * 16 + m;
                #pragma unroll
                for (int r = 0; r < 8; ++r) {
                    const int row = r + 8 * h;
                    h2buf[row * 64 + col] =
                        (_Float16)tanh_scaled(fmaf(c2[nt][r], TWO_LOG2E, b2s[nt]));
                }
            }
            asm volatile("s_wait_dscnt 0x0" ::: "memory");

            // ---- re-layout h2 into A-fragments via 128-bit LDS loads.
            // Row-major 16-bit A tile: slots 0..7 = 8 consecutive K at (m*stride + 8h),
            // slots 8..15 at (m*stride + 16 + 8h). All 16B aligned (row stride = 128 B).
            v16h a3[2];
            #pragma unroll
            for (int kt = 0; kt < 2; ++kt) {
                const _Float16* base = &h2buf[m * 64 + kt * 32 + 8 * h];
                v8h lo = *(const v8h*)(base);       // K = kt*32 + 8h + 0..7
                v8h hi = *(const v8h*)(base + 16);  // K = kt*32 + 16 + 8h + 0..7
                #pragma unroll
                for (int i = 0; i < 8; ++i) {
                    a3[kt][i]     = lo[i];
                    a3[kt][i + 8] = hi[i];
                }
            }

            // ---- layer3: (16x64) x (64x10 padded to 16) -> 2 chained WMMAs
            v8f c3 = {};
            c3 = __builtin_amdgcn_wmma_f32_16x16x32_f16(
                false, a3[0], false, b3frag[0], (short)0, c3, false, false);
            c3 = __builtin_amdgcn_wmma_f32_16x16x32_f16(
                false, a3[1], false, b3frag[1], (short)0, c3, false, false);

            // sigmoid (one FMA of prep) + scatter s[n][row][col] (cols 0..9 only)
            if (m < 10) {
                #pragma unroll
                for (int r = 0; r < 8; ++r) {
                    const int row = r + 8 * h;
                    sbuf[n][row * 10 + m] =
                        sigmoid_scaled(fmaf(c3[r], -LOG2E, b3s));
                }
            }
        }
        asm volatile("s_wait_dscnt 0x0" ::: "memory");

        // ---- cyclic-permanent combine: lane handles (row=m, c=h)
        {
            float y = 0.0f;
            #pragma unroll
            for (int p = 0; p < NPER; ++p) {
                float prod = 1.0f;
                #pragma unroll
                for (int q = 0; q < NPER; ++q) {
                    int r = p + q; if (r >= NPER) r -= NPER;       // PERM[p,q] = (p+q)%5
                    prod *= sbuf[q][m * 10 + h * 5 + r];
                }
                y += prod;
            }
            ybuf[m * 2 + h] = y;
        }
        asm volatile("s_wait_dscnt 0x0" ::: "memory");

        // ---- head MLP 2 -> 64 -> 1 (one lane per row)
        if (h == 0) {
            const float y0 = ybuf[m * 2 + 0];
            const float y1 = ybuf[m * 2 + 1];
            float acc = 0.0f;
            #pragma unroll 8
            for (int f = 0; f < 64; ++f) {
                float hh = fmaf(y0, Wfc[f], y1 * Wfc[64 + f]);
                hh = fmaxf(hh, 0.0f);
                acc = fmaf(hh, Wout[f], acc);
            }
            out[rowBase + m] = acc + bout[0];
        }
    }
}

extern "C" void kernel_launch(void* const* d_in, const int* in_sizes, int n_in,
                              void* d_out, int out_size, void* d_ws, size_t ws_size,
                              hipStream_t stream) {
    const float* inputs = (const float*)d_in[0];
    const float* W1     = (const float*)d_in[1];
    const float* b1     = (const float*)d_in[2];
    const float* W2     = (const float*)d_in[3];
    const float* b2     = (const float*)d_in[4];
    const float* W3     = (const float*)d_in[5];
    const float* b3     = (const float*)d_in[6];
    const float* Wfc    = (const float*)d_in[7];
    const float* Wout   = (const float*)d_in[8];
    const float* bout   = (const float*)d_in[9];
    float* out          = (float*)d_out;

    group_invariance_fused<<<NUM_BLOCKS, BLOCK_THREADS, 0, stream>>>(
        inputs, W1, b1, W2, b2, W3, b3, Wfc, Wout, bout, out);
}
